// BigBirdCCREAttention_13511967113467
// MI455X (gfx1250) — compile-verified, matrix operand order checked by reading, not run on hardware
//
#include <hip/hip_runtime.h>

typedef __attribute__((ext_vector_type(16))) __bf16 v16bf;
typedef __attribute__((ext_vector_type(8)))  float  v8f;
typedef unsigned short ushort_t;

// ---------- helpers ----------
__device__ __forceinline__ unsigned short f32_to_bf16(float f) {
  unsigned int x = __float_as_uint(f);
  x += 0x7fffu + ((x >> 16) & 1u);   // round-to-nearest-even
  return (unsigned short)(x >> 16);
}
union BFU { unsigned short u; __bf16 b; };
__device__ __forceinline__ __bf16 us2bf(unsigned short u) { BFU t; t.u = u; return t.b; }

__device__ __forceinline__ v8f wmma_bf16(v16bf a, v16bf b, v8f c) {
  // D(f32,16x16) = A(bf16,16x32) * B(bf16,32x16) + C
  return __builtin_amdgcn_wmma_f32_16x16x32_bf16(false, a, false, b, (short)0, c,
                                                 false, false);
}

// ---------- conversion kernels ----------
__global__ void k_cvt_bf16(const float* __restrict__ src, ushort_t* __restrict__ dst, int n) {
  int i = blockIdx.x * blockDim.x + threadIdx.x;
  if (i < n) dst[i] = f32_to_bf16(src[i]);
}

// convert with zero tail padding (so GEMM M can be padded to a multiple of 32)
__global__ void k_cvt_bf16_pad(const float* __restrict__ src, ushort_t* __restrict__ dst,
                               int n_src, int n_tot) {
  int i = blockIdx.x * blockDim.x + threadIdx.x;
  if (i < n_tot) dst[i] = (i < n_src) ? f32_to_bf16(src[i]) : (ushort_t)0;
}

// src [K,N] row-major f32 -> dst [N,K] row-major bf16 (transpose)
__global__ void k_cvt_bf16_t(const float* __restrict__ src, ushort_t* __restrict__ dst,
                             int K, int N) {
  int i = blockIdx.x * blockDim.x + threadIdx.x;
  if (i < K * N) {
    int k = i / N, n = i % N;
    dst[(size_t)n * K + k] = f32_to_bf16(src[i]);
  }
}

// ---------- WMMA GEMM: C[M,N] = A[M,K](bf16) * Bt[N,K](bf16)^T (+bias) ----------
// One wave per 32(M)x64(N) block: 8 accumulators, A frags reused 4x, B frags 2x.
// Requires M % 32 == 0, N % 64 == 0, K % 32 == 0 (all callers padded accordingly).
__global__ void k_wmma_gemm(const ushort_t* __restrict__ A, const ushort_t* __restrict__ Bt,
                            float* __restrict__ C, const float* __restrict__ bias,
                            int M, int N, int K) {
  const int lane = threadIdx.x & 31;
  const int half = lane >> 4, l16 = lane & 15;
  const int col0 = blockIdx.x * 64;
  const int row0 = blockIdx.y * 32;

  const ushort_t* Ar0 = A + (size_t)(row0 + l16) * K + half * 16;
  const ushort_t* Ar1 = A + (size_t)(row0 + 16 + l16) * K + half * 16;
  const ushort_t* Bc[4];
#pragma unroll
  for (int c = 0; c < 4; c++) Bc[c] = Bt + (size_t)(col0 + c * 16 + l16) * K + half * 16;

  const v8f vzero = {0.f, 0.f, 0.f, 0.f, 0.f, 0.f, 0.f, 0.f};
  v8f acc[2][4];
#pragma unroll
  for (int rt = 0; rt < 2; rt++)
#pragma unroll
    for (int c = 0; c < 4; c++) acc[rt][c] = vzero;

  for (int k = 0; k < K; k += 32) {
    v16bf a0 = *(const v16bf*)(Ar0 + k);
    v16bf a1 = *(const v16bf*)(Ar1 + k);
#pragma unroll
    for (int c = 0; c < 4; c++) {
      v16bf bf = *(const v16bf*)(Bc[c] + k);
      acc[0][c] = wmma_bf16(a0, bf, acc[0][c]);
      acc[1][c] = wmma_bf16(a1, bf, acc[1][c]);
    }
  }

#pragma unroll
  for (int c = 0; c < 4; c++) {
    int col = col0 + c * 16 + l16;
    float bv = bias ? bias[col] : 0.f;
#pragma unroll
    for (int rt = 0; rt < 2; rt++) {
#pragma unroll
      for (int r = 0; r < 8; r++) {
        int row = row0 + rt * 16 + r + half * 8;  // C layout: vgpr r -> M = r + 8*half
        C[(size_t)row * N + col] = acc[rt][c][r] + bv;
      }
    }
  }
}

// ---------- reformat QKV: scale, add biases, per-head bf16 layouts ----------
// Qr/Kr/Vr: [B*N, H*64] f32.  Qc/Qp/Kb: [B*H, N, 64] bf16.  Vt: [B*H, 64, N] bf16.
__global__ void k_reform_qkv(const float* __restrict__ Qr, const float* __restrict__ Kr,
                             const float* __restrict__ Vr, const float* __restrict__ rcb,
                             const float* __restrict__ rpb, ushort_t* __restrict__ Qc,
                             ushort_t* __restrict__ Qp, ushort_t* __restrict__ Kb,
                             ushort_t* __restrict__ Vt) {
  int idx = blockIdx.x * blockDim.x + threadIdx.x;
  if (idx >= 2 * 1536 * 512) return;
  int d = idx & 63;
  int h = (idx >> 6) & 7;
  int n = (idx >> 9) % 1536;
  int b = idx / (1536 * 512);
  size_t src = (size_t)(b * 1536 + n) * 512 + h * 64 + d;
  size_t bh = (size_t)b * 8 + h;
  size_t dst = (bh * 1536 + n) * 64 + d;
  float q = Qr[src] * 0.125f;                 // dk^-0.5, dk=64
  Qc[dst] = f32_to_bf16(q + rcb[h * 64 + d]);
  Qp[dst] = f32_to_bf16(q + rpb[h * 64 + d]);
  Kb[dst] = f32_to_bf16(Kr[src]);
  Vt[(bh * 64 + d) * 1536 + n] = f32_to_bf16(Vr[src]);
}

// rel raw [3072(padded), H*64] f32 -> Rk [H, 3072(padded), 64] bf16 (rows < 3071 valid)
__global__ void k_reform_rel(const float* __restrict__ Rr, ushort_t* __restrict__ Rk) {
  int idx = blockIdx.x * blockDim.x + threadIdx.x;
  if (idx >= 8 * 3071 * 64) return;
  int d = idx & 63;
  int p = (idx >> 6) % 3071;
  int h = idx / (3071 * 64);
  Rk[((size_t)h * 3072 + p) * 64 + d] = f32_to_bf16(Rr[(size_t)p * 512 + h * 64 + d]);
}

// ---------- fused attention (flash-style, banded rel logits, BigBird tile skip) ----------
// grid: (B*H, N/64); block 128 = 4 waves; each wave owns 16 query rows.
#define AN 1536
__global__ __launch_bounds__(128) void k_attn(
    const ushort_t* __restrict__ Qc, const ushort_t* __restrict__ Qp,
    const ushort_t* __restrict__ Kb, const ushort_t* __restrict__ Vt,
    const ushort_t* __restrict__ Rk, ushort_t* __restrict__ Oc) {
  __shared__ float    ldsD[4][16 * 49];   // rel-logit band, padded stride
  __shared__ ushort_t ldsP[4][16 * 32];   // bf16 probabilities (C-layout -> A-layout)

  const int wave = threadIdx.x >> 5, lane = threadIdx.x & 31;
  const int half = lane >> 4, l16 = lane & 15;
  const int bh = blockIdx.x;                 // 0..15
  const int b = bh >> 3, h = bh & 7;
  const int i0 = (blockIdx.y * 4 + wave) * 16;

  const ushort_t* Qc_bh = Qc + (size_t)bh * AN * 64;
  const ushort_t* Qp_bh = Qp + (size_t)bh * AN * 64;
  const ushort_t* Kb_bh = Kb + (size_t)bh * AN * 64;
  const ushort_t* Vt_bh = Vt + (size_t)bh * 64 * AN;
  const ushort_t* Rk_h  = Rk + (size_t)h * 3072 * 64;
  float*    Dw = ldsD[wave];
  ushort_t* Pw = ldsP[wave];

  // query fragments stay resident in VGPRs
  v16bf aQc[2], aQp[2];
#pragma unroll
  for (int s = 0; s < 2; s++) {
    aQc[s] = *(const v16bf*)(Qc_bh + (size_t)(i0 + l16) * 64 + s * 32 + half * 16);
    aQp[s] = *(const v16bf*)(Qp_bh + (size_t)(i0 + l16) * 64 + s * 32 + half * 16);
  }

  const v8f vzero = {0.f, 0.f, 0.f, 0.f, 0.f, 0.f, 0.f, 0.f};
  v8f acc[4];
  float Mr[8], Lr[8];
#pragma unroll
  for (int t = 0; t < 4; t++) acc[t] = vzero;
#pragma unroll
  for (int r = 0; r < 8; r++) { Mr[r] = -3.0e38f; Lr[r] = 0.f; }

  // BigBird tile schedule (wave-uniform): the 16 query rows share one 128-block.
  // Wave i0==0 holds the 4 global rows -> needs all key tiles. Other waves need
  // only the global-column tile j0=0 plus the 3-block local window.
  const bool full = (i0 == 0);
  const int bi0 = i0 >> 7;
  const int jlo = (bi0 - 1) < 0 ? 0 : (bi0 - 1) * 128;
  const int jhi = (bi0 + 2) * 128 > AN ? AN : (bi0 + 2) * 128;

  for (int phase = 0; phase < 2; phase++) {
    int js, je;
    if (full) {
      if (phase) break;
      js = 0; je = AN;
    } else if (phase == 0) {
      js = 0; je = 32;                       // global columns (j < 4 live here)
    } else {
      js = (jlo < 32) ? 32 : jlo; je = jhi;  // local window, minus already-done tile
    }

    for (int j0 = js; j0 < je; j0 += 32) {
      // ---- content logits: (q*scale + content_bias) . k ----
      v8f c0 = vzero, c1 = vzero;
#pragma unroll
      for (int s = 0; s < 2; s++) {
        v16bf b0 = *(const v16bf*)(Kb_bh + (size_t)(j0 + l16) * 64 + s * 32 + half * 16);
        v16bf b1 = *(const v16bf*)(Kb_bh + (size_t)(j0 + 16 + l16) * 64 + s * 32 + half * 16);
        c0 = wmma_bf16(aQc[s], b0, c0);
        c1 = wmma_bf16(aQc[s], b1, c1);
      }

      // ---- rel-logit band: D[i,c] = (q+pos_bias)_i . relK[(N-1)+j0-i0-15+c] ----
      const int base = 1535 + j0 - i0 - 15;
#pragma unroll
      for (int t = 0; t < 3; t++) {
        v8f d = vzero;
        int krow = base + t * 16 + l16;
        krow = min(max(krow, 0), 3070);      // pad column c=47 clamped (never gathered)
        const ushort_t* rp = Rk_h + (size_t)krow * 64;
#pragma unroll
        for (int s = 0; s < 2; s++) {
          v16bf bb = *(const v16bf*)(rp + s * 32 + half * 16);
          d = wmma_bf16(aQp[s], bb, d);
        }
#pragma unroll
        for (int r = 0; r < 8; r++) {
          int m = r + half * 8;
          Dw[m * 49 + t * 16 + l16] = d[r];
        }
      }

      // ---- gather band + BigBird mask ----
      float S0[8], S1[8];
#pragma unroll
      for (int r = 0; r < 8; r++) {
        int m = r + half * 8;
        int i = i0 + m;
        int ja = j0 + l16, jb = j0 + 16 + l16;
        float sa = c0[r] + Dw[m * 49 + (l16 - m + 15)];
        float sb = c1[r] + Dw[m * 49 + (l16 + 16 - m + 15)];
        int bi = i >> 7;                      // block size 128
        bool oka = (abs(bi - (ja >> 7)) <= 1) || (i < 4) || (ja < 4);
        bool okb = (abs(bi - (jb >> 7)) <= 1) || (i < 4) || (jb < 4);
        S0[r] = oka ? sa : -1.0e30f;
        S1[r] = okb ? sb : -1.0e30f;
      }

      // ---- online softmax (per query row, reduce across 16 lanes of the half) ----
#pragma unroll
      for (int r = 0; r < 8; r++) {
        float tm = fmaxf(S0[r], S1[r]);
        tm = fmaxf(tm, __shfl_xor(tm, 1));
        tm = fmaxf(tm, __shfl_xor(tm, 2));
        tm = fmaxf(tm, __shfl_xor(tm, 4));
        tm = fmaxf(tm, __shfl_xor(tm, 8));
        float nm = fmaxf(Mr[r], tm);
        float corr = __expf(Mr[r] - nm);
        float p0 = __expf(S0[r] - nm);
        float p1 = __expf(S1[r] - nm);
        float rs = p0 + p1;
        rs += __shfl_xor(rs, 1);
        rs += __shfl_xor(rs, 2);
        rs += __shfl_xor(rs, 4);
        rs += __shfl_xor(rs, 8);
        Lr[r] = Lr[r] * corr + rs;
        Mr[r] = nm;
#pragma unroll
        for (int t = 0; t < 4; t++) acc[t][r] *= corr;
        int m = r + half * 8;
        Pw[m * 32 + l16] = f32_to_bf16(p0);
        Pw[m * 32 + 16 + l16] = f32_to_bf16(p1);
      }

      // ---- P @ V : A = P[16x32] (via LDS transpose), B = Vt slice [32k x 16dv] ----
      v16bf aP = *(const v16bf*)(Pw + l16 * 32 + half * 16);
#pragma unroll
      for (int t = 0; t < 4; t++) {
        v16bf bv = *(const v16bf*)(Vt_bh + (size_t)(t * 16 + l16) * AN + j0 + half * 16);
        acc[t] = wmma_bf16(aP, bv, acc[t]);
      }
    }
  }

  // ---- normalize and store to [B*N, H*64] bf16 for the output projection ----
#pragma unroll
  for (int t = 0; t < 4; t++) {
#pragma unroll
    for (int r = 0; r < 8; r++) {
      int m = r + half * 8;
      int i = i0 + m;
      int dv = t * 16 + l16;
      float v = acc[t][r] / Lr[r];
      Oc[((size_t)(b * AN + i)) * 512 + h * 64 + dv] = f32_to_bf16(v);
    }
  }
}

// ---------- host ----------
static inline int cdiv(long long a, long long b) { return (int)((a + b - 1) / b); }

extern "C" void kernel_launch(void* const* d_in, const int* in_sizes, int n_in,
                              void* d_out, int out_size, void* d_ws, size_t ws_size,
                              hipStream_t stream) {
  const float* x    = (const float*)d_in[0];   // [2,1536,1536]
  const float* Wq   = (const float*)d_in[1];   // [1536,512]
  const float* Wk   = (const float*)d_in[2];
  const float* Wv   = (const float*)d_in[3];
  const float* Wrel = (const float*)d_in[4];   // [192,512]
  const float* rcb  = (const float*)d_in[5];   // [1,8,1,64]
  const float* rpb  = (const float*)d_in[6];
  const float* Wo   = (const float*)d_in[7];   // [512,1536]
  const float* bo   = (const float*)d_in[8];   // [1536]
  const float* pos  = (const float*)d_in[9];   // [3071,192]
  float* out = (float*)d_out;

  char* ws = (char*)d_ws;
  size_t off = 0;
  auto carve = [&](size_t bytes) -> void* {
    void* p = ws + off;
    off = (off + bytes + 255) & ~(size_t)255;
    return p;
  };
  ushort_t* xb    = (ushort_t*)carve(3072ull * 1536 * 2);
  ushort_t* posb  = (ushort_t*)carve(3072ull * 192 * 2);     // padded to 3072 rows
  ushort_t* Wqt   = (ushort_t*)carve(512ull * 1536 * 2);
  ushort_t* Wkt   = (ushort_t*)carve(512ull * 1536 * 2);
  ushort_t* Wvt   = (ushort_t*)carve(512ull * 1536 * 2);
  ushort_t* Wrelt = (ushort_t*)carve(512ull * 192 * 2);
  ushort_t* Wot   = (ushort_t*)carve(1536ull * 512 * 2);
  float*    Qraw  = (float*)carve(3072ull * 512 * 4);
  float*    Kraw  = (float*)carve(3072ull * 512 * 4);
  float*    Vraw  = (float*)carve(3072ull * 512 * 4);
  float*    Rraw  = (float*)carve(3072ull * 512 * 4);
  ushort_t* Qc    = (ushort_t*)carve(16ull * 1536 * 64 * 2);
  ushort_t* Qp    = (ushort_t*)carve(16ull * 1536 * 64 * 2);
  ushort_t* Kb    = (ushort_t*)carve(16ull * 1536 * 64 * 2);
  ushort_t* Vt    = (ushort_t*)carve(16ull * 64 * 1536 * 2);
  ushort_t* Rk    = (ushort_t*)carve(8ull * 3072 * 64 * 2);
  ushort_t* Oc    = (ushort_t*)carve(3072ull * 512 * 2);

  const int T = 256;
  // conversions
  k_cvt_bf16<<<cdiv(3072LL * 1536, T), T, 0, stream>>>(x, xb, 3072 * 1536);
  k_cvt_bf16_pad<<<cdiv(3072LL * 192, T), T, 0, stream>>>(pos, posb, 3071 * 192,
                                                          3072 * 192);
  k_cvt_bf16_t<<<cdiv(1536LL * 512, T), T, 0, stream>>>(Wq, Wqt, 1536, 512);
  k_cvt_bf16_t<<<cdiv(1536LL * 512, T), T, 0, stream>>>(Wk, Wkt, 1536, 512);
  k_cvt_bf16_t<<<cdiv(1536LL * 512, T), T, 0, stream>>>(Wv, Wvt, 1536, 512);
  k_cvt_bf16_t<<<cdiv(192LL * 512, T), T, 0, stream>>>(Wrel, Wrelt, 192, 512);
  k_cvt_bf16_t<<<cdiv(512LL * 1536, T), T, 0, stream>>>(Wo, Wot, 512, 1536);

  // projections (all shapes padded: guard-free WMMA inner loops)
  dim3 gq(512 / 64, 3072 / 32);
  k_wmma_gemm<<<gq, 32, 0, stream>>>(xb, Wqt, Qraw, nullptr, 3072, 512, 1536);
  k_wmma_gemm<<<gq, 32, 0, stream>>>(xb, Wkt, Kraw, nullptr, 3072, 512, 1536);
  k_wmma_gemm<<<gq, 32, 0, stream>>>(xb, Wvt, Vraw, nullptr, 3072, 512, 1536);
  k_wmma_gemm<<<gq, 32, 0, stream>>>(posb, Wrelt, Rraw, nullptr, 3072, 512, 192);

  // reformats
  k_reform_qkv<<<cdiv(2LL * 1536 * 512, T), T, 0, stream>>>(Qraw, Kraw, Vraw, rcb, rpb,
                                                            Qc, Qp, Kb, Vt);
  k_reform_rel<<<cdiv(8LL * 3071 * 64, T), T, 0, stream>>>(Rraw, Rk);

  // fused attention
  dim3 ga(16, 1536 / 64);
  k_attn<<<ga, 128, 0, stream>>>(Qc, Qp, Kb, Vt, Rk, Oc);

  // output projection + bias
  dim3 go(1536 / 64, 3072 / 32);
  k_wmma_gemm<<<go, 32, 0, stream>>>(Oc, Wot, out, bo, 3072, 1536, 512);
}
